// AxialAttention_120259084317
// MI455X (gfx1250) — compile-verified, matrix operand order checked by reading, not run on hardware
//
#include <hip/hip_runtime.h>
#include <hip/hip_fp16.h>
#include <stdint.h>

typedef __attribute__((ext_vector_type(16))) _Float16 v16h;
typedef __attribute__((ext_vector_type(8)))  _Float16 v8h;
typedef __attribute__((ext_vector_type(8)))  float    v8f;
typedef __attribute__((ext_vector_type(4)))  unsigned u32x4;
typedef __attribute__((ext_vector_type(4)))  int      i32x4;
typedef __attribute__((ext_vector_type(8)))  int      i32x8;

// Problem constants
constexpr int cN   = 16;
constexpr int cCIN = 128;
constexpr int cOUT = 128;
constexpr int cG   = 8;
constexpr int cK   = 56;          // H == W == 56
constexpr int cGP  = cOUT / cG;   // 16
constexpr int cHP  = cGP / 2;     // 8
constexpr int cB   = cN * cK;     // 896 batched columns
constexpr int cQKV = 2 * cOUT;    // 256 channels
constexpr float cEPS = 1e-5f;

// Workspace byte offsets
constexpr size_t OFF_WQ   = 0;                       // 256*128 f16      = 65536 B
constexpr size_t OFF_EMB  = 65536;                   // 32*56*64 f16     = 229376 B
constexpr size_t OFF_BIAS = OFF_EMB + 229376;        // 256 f32
constexpr size_t OFF_SIMS = OFF_BIAS + 1024;
constexpr size_t OFF_SIMB = OFF_SIMS + 128;
constexpr size_t OFF_OUTS = OFF_SIMB + 128;
constexpr size_t OFF_OUTB = OFF_OUTS + 1024;
constexpr size_t OFF_QKV  = 1u << 20;                // 896*256*64 f16 = 29,360,128 B

// Assemble a 16-half WMMA fragment from two contiguous 8-half runs (-> ds/global b128)
__device__ __forceinline__ v16h frag16(const _Float16* lo, const _Float16* hi) {
    v8h l = *(const v8h*)lo;
    v8h h = *(const v8h*)hi;
    return __builtin_shufflevector(l, h, 0,1,2,3,4,5,6,7,8,9,10,11,12,13,14,15);
}

__device__ __forceinline__ unsigned lds_addr32(const void* p) {
    return (unsigned)(uintptr_t)p;   // low 32 bits of generic LDS addr == LDS offset
}

// ---------------------------------------------------------------------------
// Kernel 1: fold BN into W (f16), biases, BN scale/shift tables, rel-embedding
// ---------------------------------------------------------------------------
__global__ __launch_bounds__(256) void prep_kernel(
    const float* __restrict__ w_qkv, const float* __restrict__ relative,
    const float* __restrict__ qg, const float* __restrict__ qb,
    const float* __restrict__ qm, const float* __restrict__ qv,
    const float* __restrict__ sg, const float* __restrict__ sb,
    const float* __restrict__ sm, const float* __restrict__ svv,
    const float* __restrict__ og, const float* __restrict__ ob,
    const float* __restrict__ om, const float* __restrict__ ov,
    _Float16* __restrict__ wq16, _Float16* __restrict__ emb16,
    float* __restrict__ bias, float* __restrict__ sims, float* __restrict__ simb,
    float* __restrict__ outs, float* __restrict__ outb)
{
    const int t = blockIdx.x * blockDim.x + threadIdx.x;
    const int nthr = gridDim.x * blockDim.x;

    for (int idx = t; idx < cQKV * cCIN; idx += nthr) {
        const int o = idx / cCIN;
        const float s = qg[o] * rsqrtf(qv[o] + cEPS);
        wq16[idx] = (_Float16)(w_qkv[idx] * s);
    }
    for (int o = t; o < cQKV; o += nthr) {
        const float s = qg[o] * rsqrtf(qv[o] + cEPS);
        bias[o] = qb[o] - qm[o] * s;
    }
    // emb16[c][i][j] = relative[c][i - j + K - 1], j padded to 64 with zeros
    for (int idx = t; idx < 32 * cK * 64; idx += nthr) {
        const int j = idx & 63;
        const int rest = idx >> 6;
        const int i = rest % cK;
        const int c = rest / cK;
        float val = 0.f;
        if (j < cK) val = relative[c * (2 * cK - 1) + (i - j + cK - 1)];
        emb16[idx] = (_Float16)val;
    }
    for (int ch = t; ch < 3 * cG; ch += nthr) {
        const float s = sg[ch] * rsqrtf(svv[ch] + cEPS);
        sims[ch] = s; simb[ch] = sb[ch] - sm[ch] * s;
    }
    for (int ch = t; ch < cQKV; ch += nthr) {
        const float s = og[ch] * rsqrtf(ov[ch] + cEPS);
        outs[ch] = s; outb[ch] = ob[ch] - om[ch] * s;
    }
}

// ---------------------------------------------------------------------------
// Kernel 2: QKV projection. 1 block per b, 8 waves, WMMA f16 -> f32.
// x-slice staged via async global->LDS (f32), converted to K-contiguous f16.
// ---------------------------------------------------------------------------
__global__ __launch_bounds__(256) void qkv_kernel(
    const float* __restrict__ x, const _Float16* __restrict__ wq16,
    const float* __restrict__ bias, _Float16* __restrict__ qkv16)
{
    __shared__ float    xraw[cCIN][cK];   // 28 KB, async-staged f32 slice
    __shared__ _Float16 xsT[64][cCIN];    // 16 KB, transposed (h, c): K-contiguous

    const int b = blockIdx.x;
    const int n = b / cK, w = b % cK;
    const int tid = threadIdx.x;

    __builtin_prefetch(wq16, 0, 0);       // global_prefetch_b8: W is L2-resident

    // Async-copy the strided x[:, :, :, w] gather into LDS (ASYNCcnt path)
    for (int idx = tid; idx < cCIN * cK; idx += 256) {
        const int c = idx / cK, h = idx % cK;
        const unsigned lo = lds_addr32(&xraw[c][h]);
        const unsigned long long ga =
            (unsigned long long)(uintptr_t)&x[((size_t)(n * cCIN + c) * cK + h) * cK + w];
        asm volatile("global_load_async_to_lds_b32 %0, %1, off"
                     :: "v"(lo), "v"(ga) : "memory");
    }
    asm volatile("s_wait_asynccnt 0x0" ::: "memory");
    __syncthreads();

    // Convert + transpose: xsT[h][c] = f16(xraw[c][h]), h padded to 64
    for (int idx = tid; idx < 64 * cCIN; idx += 256) {
        const int h = idx / cCIN, c = idx % cCIN;
        xsT[h][c] = (_Float16)((h < cK) ? xraw[c][h] : 0.f);
    }
    __syncthreads();

    const int wave = tid >> 5, lane = tid & 31;
    const int col   = lane & 15;
    const int khalf = (lane < 16) ? 0 : 8;
    const int mofs  = khalf;

    for (int mt = wave * 2; mt < wave * 2 + 2; ++mt) {
        const _Float16* wrow = wq16 + (size_t)(mt * 16 + col) * cCIN;
        for (int nt = 0; nt < 4; ++nt) {
            const _Float16* xrow = &xsT[nt * 16 + col][0];
            v8f acc = {};
            #pragma unroll
            for (int kc = 0; kc < 4; ++kc) {
                const int kb = kc * 32 + khalf;
                const v16h a  = frag16(wrow + kb, wrow + kb + 16);
                const v16h bf = frag16(xrow + kb, xrow + kb + 16);
                acc = __builtin_amdgcn_wmma_f32_16x16x32_f16(
                        false, a, false, bf, (short)0, acc, false, false);
            }
            #pragma unroll
            for (int r = 0; r < 8; ++r) {
                const int m = mt * 16 + r + mofs;
                const int h = nt * 16 + col;
                qkv16[((size_t)b * cQKV + m) * 64 + h] = (_Float16)(acc[r] + bias[m]);
            }
        }
    }
}

// ---------------------------------------------------------------------------
// Kernel 3: fused attention per (b,g). 4 waves.
//  q/k/v tile staged via TDM tensor_load_to_lds (2D tile 32x128B, contiguous)
//  logits VALU -> softmax -> SV via WMMA -> sve VALU -> fused out-BN store
// ---------------------------------------------------------------------------
__global__ __launch_bounds__(128) void attn_kernel(
    const _Float16* __restrict__ qkv16, const _Float16* __restrict__ emb16,
    const float* __restrict__ sims, const float* __restrict__ simb,
    const float* __restrict__ outs, const float* __restrict__ outb,
    float* __restrict__ out)
{
    __shared__ _Float16 qkvs[32][64];    // q: rows 0-7, k: 8-15, v: 16-31
    __shared__ float    simf[cK][64];    // logits, then probabilities (f32)
    __shared__ _Float16 sim16[64][64];   // probabilities (f16), zero-padded
    __shared__ float    svb[cGP][64];    // SV result

    const int blk = blockIdx.x;
    const int b = blk / cG, g = blk % cG;
    const int n = b / cK, w = b % cK;
    const int tid = threadIdx.x;

    // --- Stage q/k/v (32 rows x 64 halves, contiguous) via Tensor Data Mover
#if __has_builtin(__builtin_amdgcn_tensor_load_to_lds)
    if (tid < 32) {   // wave 0 issues one TDM op; EXEC ignored by TDM
        const unsigned long long ga =
            (unsigned long long)(uintptr_t)(qkv16 + ((size_t)b * cQKV + g * 32) * 64);
        const unsigned ldsb = lds_addr32(&qkvs[0][0]);
        // D# group0: count=1 | lds_addr | global_addr[56:0] | type=2
        u32x4 g0 = { 1u, ldsb, (unsigned)ga,
                     (unsigned)((ga >> 32) & 0x1FFFFFFu) | (2u << 30) };
        // D# group1: data_size=2B; tensor_dim0=64, tensor_dim1=32;
        //            tile_dim0=64, tile_dim1=32; tensor_dim0_stride=64
        i32x8 g1 = { (int)(1u << 16), (int)(64u << 16), (int)(32u << 16),
                     (int)(64u << 16), 32, 64, 0, 0 };
        i32x4 z4 = { 0, 0, 0, 0 };
#if __clang_major__ >= 23
        i32x8 z8 = { 0, 0, 0, 0, 0, 0, 0, 0 };
        __builtin_amdgcn_tensor_load_to_lds(g0, g1, z4, z4, z8, 0);
#else
        __builtin_amdgcn_tensor_load_to_lds(g0, g1, z4, z4, 0);
#endif
        __builtin_amdgcn_s_wait_tensorcnt(0);
    }
#else
    for (int idx = tid; idx < 32 * 64; idx += 128)
        qkvs[idx >> 6][idx & 63] =
            qkv16[((size_t)b * cQKV + g * 32 + (idx >> 6)) * 64 + (idx & 63)];
#endif
    // Zero sim16 padding while the TDM runs
    for (int idx = tid; idx < 64 * 64; idx += 128)
        sim16[idx >> 6][idx & 63] = (_Float16)0.f;
    __syncthreads();

    const float s0 = sims[g],          h0 = simb[g];
    const float s1 = sims[cG + g],     h1 = simb[cG + g];
    const float s2 = sims[2 * cG + g], h2 = simb[2 * cG + g];

    // Logits: qk + qr + kr^T with BN folded
    for (int idx = tid; idx < cK * cK; idx += 128) {
        const int i = idx / cK, j = idx % cK;
        float qk = 0.f, qr = 0.f, kr = 0.f;
        #pragma unroll
        for (int c = 0; c < cHP; ++c) {
            const float qv_ = (float)qkvs[c][i];
            const float kv_ = (float)qkvs[cHP + c][j];
            qk += qv_ * kv_;
            qr += qv_ * (float)emb16[((size_t)c * cK + i) * 64 + j];           // q_emb
            kr += kv_ * (float)emb16[((size_t)(cHP + c) * cK + j) * 64 + i];   // k_emb^T
        }
        simf[i][j] = (s0 * qk + h0) + (s1 * qr + h1) + (s2 * kr + h2);
    }
    __syncthreads();

    // Rowwise softmax; write probs to simf (f32) and sim16 (f16, zero-padded)
    if (tid < cK) {
        const int i = tid;
        float mx = -1e30f;
        for (int j = 0; j < cK; ++j) mx = fmaxf(mx, simf[i][j]);
        float sum = 0.f;
        for (int j = 0; j < cK; ++j) { const float e = __expf(simf[i][j] - mx); simf[i][j] = e; sum += e; }
        const float inv = 1.f / sum;
        for (int j = 0; j < cK; ++j) { const float p = simf[i][j] * inv; simf[i][j] = p; sim16[i][j] = (_Float16)p; }
        for (int j = cK; j < 64; ++j) sim16[i][j] = (_Float16)0.f;
    }
    __syncthreads();

    // SV: sv[c][i] = sum_j v[c][j] * sim[i][j]  -- WMMA, M=c(16) N=i K=j(64)
    {
        const int wave = tid >> 5, lane = tid & 31;
        const int col   = lane & 15;
        const int khalf = (lane < 16) ? 0 : 8;
        const int it = wave;                          // i tile 0..3
        const _Float16* vrow = &qkvs[16 + col][0];
        const _Float16* srow = &sim16[it * 16 + col][0];
        v8f acc = {};
        #pragma unroll
        for (int kc = 0; kc < 2; ++kc) {
            const int jb = kc * 32 + khalf;
            const v16h a  = frag16(vrow + jb, vrow + jb + 16);
            const v16h bf = frag16(srow + jb, srow + jb + 16);
            acc = __builtin_amdgcn_wmma_f32_16x16x32_f16(
                    false, a, false, bf, (short)0, acc, false, false);
        }
        #pragma unroll
        for (int r = 0; r < 8; ++r)
            svb[r + khalf][it * 16 + col] = acc[r];
    }
    __syncthreads();

    // sve + fused output BN + pair-sum, direct store
    for (int idx = tid; idx < cGP * cK; idx += 128) {
        const int c = idx / cK, i = idx % cK;
        float sve = 0.f;
        for (int j = 0; j < cK; ++j)
            sve += simf[i][j] * (float)emb16[((size_t)(cGP + c) * cK + i) * 64 + j];  // v_emb
        const int ch0 = g * 32 + 2 * c;
        const int ch1 = ch0 + 1;
        const float val = outs[ch0] * svb[c][i] + outb[ch0]
                        + outs[ch1] * sve      + outb[ch1];
        const int oc = g * cGP + c;
        out[((size_t)(n * cOUT + oc) * cK + i) * cK + w] = val;
    }
}

// ---------------------------------------------------------------------------
extern "C" void kernel_launch(void* const* d_in, const int* in_sizes, int n_in,
                              void* d_out, int out_size, void* d_ws, size_t ws_size,
                              hipStream_t stream) {
    const float* x        = (const float*)d_in[0];
    const float* w_qkv    = (const float*)d_in[1];
    const float* relative = (const float*)d_in[2];
    const float* qg = (const float*)d_in[3];
    const float* qb = (const float*)d_in[4];
    const float* qm = (const float*)d_in[5];
    const float* qv = (const float*)d_in[6];
    const float* sg = (const float*)d_in[7];
    const float* sb = (const float*)d_in[8];
    const float* sm = (const float*)d_in[9];
    const float* sv = (const float*)d_in[10];
    const float* og = (const float*)d_in[11];
    const float* ob = (const float*)d_in[12];
    const float* om = (const float*)d_in[13];
    const float* ov = (const float*)d_in[14];

    char* ws = (char*)d_ws;
    _Float16* wq16  = (_Float16*)(ws + OFF_WQ);
    _Float16* emb16 = (_Float16*)(ws + OFF_EMB);
    float* bias = (float*)(ws + OFF_BIAS);
    float* sims = (float*)(ws + OFF_SIMS);
    float* simb = (float*)(ws + OFF_SIMB);
    float* outs = (float*)(ws + OFF_OUTS);
    float* outb = (float*)(ws + OFF_OUTB);
    _Float16* qkv16 = (_Float16*)(ws + OFF_QKV);

    prep_kernel<<<64, 256, 0, stream>>>(w_qkv, relative, qg, qb, qm, qv,
                                        sg, sb, sm, sv, og, ob, om, ov,
                                        wq16, emb16, bias, sims, simb, outs, outb);
    qkv_kernel<<<cB, 256, 0, stream>>>(x, wq16, bias, qkv16);
    attn_kernel<<<cB * cG, 128, 0, stream>>>(qkv16, emb16, sims, simb, outs, outb,
                                             (float*)d_out);
}
